// Seq2SeqRNN_All_72773925863724
// MI455X (gfx1250) — matrix operand order repaired and checked by reading, not compiled, not run on hardware
//
#include <hip/hip_runtime.h>
#include <math.h>

// ---------------------------------------------------------------------------
// CDNA5 WMMA (wave32): v_wmma_f32_16x16x32_bf16, A 16x32 bf16, B 32x16 bf16,
// C/D 16x16 f32 (8 VGPRs).
// ---------------------------------------------------------------------------
typedef __bf16 v16bf __attribute__((ext_vector_type(16)));
typedef float  v8f   __attribute__((ext_vector_type(8)));

#define BS 64        // batch
#define NH 256       // encoder hidden
#define ED 300       // embed dim (enc/dec), decoder hidden
#define EDP 320      // ED padded to multiple of 32 (K padding)
#define H3 768       // 3*NH
#define D3 900       // 3*ED
#define D3P 1024     // D3 padded to multiple of 128 (N padding for B tiles)
#define EDNP 384     // ED padded to multiple of 128 (N padding for B tiles)
#define VDEC 32000
#define XCATW 812    // ED + 2*NH
#define XCATWP 832   // padded to multiple of 32
#define OUT_SL 32

__device__ __forceinline__ float sigmoidf_(float x) {
  return 1.0f / (1.0f + __expf(-x));
}

__device__ __forceinline__ v8f wmma_bf16(v16bf a, v16bf b, v8f c) {
#if defined(__HIP_DEVICE_COMPILE__)
  return __builtin_amdgcn_wmma_f32_16x16x32_bf16(false, a, false, b, (short)0, c,
                                                 false, false);
#else
  (void)a; (void)b; return c;
#endif
}

// A operand (16x32 bf16): lane holds row M = lane&15; VGPR q -> K pair at
// (q<4 ? 2q : 2q+8) + 8*(lane>>4).  Row-major A, stride lda (elements).
// Unguarded: caller guarantees row valid and K padded to multiple of 32.
__device__ __forceinline__ v16bf load_a_frag(const __bf16* A, int lda, int row,
                                             int kb, int lh) {
  union { v16bf v; unsigned u[8]; } r;
  const __bf16* p = A + (size_t)row * lda + kb + 8 * lh;
#pragma unroll
  for (int q = 0; q < 8; ++q) {
    int ko = (q < 4) ? 2 * q : 2 * q + 8;   // bytes 0,4,8,12 and 32,36,40,44
    __builtin_memcpy(&r.u[q], p + ko, 4);
  }
  return r.v;
}

// B operand (32x16 bf16): lane holds col N = lane&15; VGPR q -> K pair at
// (q<4 ? 2q : 2q+8) + 8*(lane>>4).  B given as N x K row-major (weight matrix), zero-padded
// in both N (to tile coverage) and K (to multiple of 32).
__device__ __forceinline__ v16bf load_b_frag(const __bf16* Bt, int ldb, int col,
                                             int kb, int lh) {
  union { v16bf v; unsigned u[8]; } r;
  const __bf16* p = Bt + (size_t)col * ldb + kb + 8 * lh;
#pragma unroll
  for (int q = 0; q < 8; ++q)
    __builtin_memcpy(&r.u[q], p + ((q < 4) ? 2 * q : 2 * q + 8), 4);
  return r.v;
}

// ---------------------------------------------------------------------------
// Branch-free WMMA GEMM: C(f32, MxN) = A(bf16, MxKp) * Bt(bf16, NpxKp)^T
// + bias(N).  Optional bf16 mirror with its own (padded) stride.
// Requirements: M % 32 == 0, Kp % 32 == 0, Bt has >= gridcover(N) rows.
// Block = 256 threads = 8 waves; each wave computes two adjacent 16x16
// N-tiles reusing its A fragment -> block tile 32(M) x 128(N).
// ---------------------------------------------------------------------------
__global__ __launch_bounds__(256) void wmma_gemm_nt(
    const __bf16* __restrict__ A, int lda,
    const __bf16* __restrict__ Bt, int ldb,
    const float* __restrict__ bias,
    float* __restrict__ C, int ldc,
    __bf16* __restrict__ Cbf, int ldcb,
    int M, int N, int Kp) {
  int wave = threadIdx.x >> 5, lane = threadIdx.x & 31;
  int lm = lane & 15, lh = lane >> 4;
  int m0 = blockIdx.y * 32 + (wave >> 2) * 16;
  int n0 = blockIdx.x * 128 + (wave & 3) * 32;
  v8f acc0 = {}, acc1 = {};
  for (int kb = 0; kb < Kp; kb += 32) {
    v16bf a  = load_a_frag(A, lda, m0 + lm, kb, lh);
    v16bf b0 = load_b_frag(Bt, ldb, n0 + lm, kb, lh);
    v16bf b1 = load_b_frag(Bt, ldb, n0 + 16 + lm, kb, lh);
    acc0 = wmma_bf16(a, b0, acc0);
    acc1 = wmma_bf16(a, b1, acc1);
  }
#pragma unroll
  for (int tile = 0; tile < 2; ++tile) {
    int n = n0 + tile * 16 + lm;
    v8f acc = tile ? acc1 : acc0;
    if (n < N) {
      float bb = bias ? bias[n] : 0.0f;
#pragma unroll
      for (int q = 0; q < 8; ++q) {
        int m = m0 + q + 8 * lh;            // M % 32 == 0 -> always valid
        float v = acc[q] + bb;
        C[(size_t)m * ldc + n] = v;
        if (Cbf) Cbf[(size_t)m * ldcb + n] = (__bf16)v;
      }
    }
  }
}

// ---------------------------------------------------------------------------
// Persistent bidirectional GRU scan (one workgroup per direction, grid=2).
// Hidden state (64 x 256) lives in LDS as bf16; per step computes
// gh = h @ Whh^T + bhh via WMMA (K=256), combines with precomputed gi
// (contains x@Wih^T + bih), applies GRU gates, updates h in LDS.
// Dynamic LDS: h (32KB) + r (32KB) + z (32KB) = 96KB (<< 320KB/WGP).
// ---------------------------------------------------------------------------
__global__ __launch_bounds__(256) void gru_scan(
    const float* __restrict__ gi,    // (2, sl, 64, 768)
    const __bf16* __restrict__ Whh,  // (2, 768, 256)
    const float* __restrict__ bhh,   // (2, 768)
    float* __restrict__ yf32,        // (sl, 64, ycols) or null
    __bf16* __restrict__ ybf16,      // (sl, 64, ycols)
    int ycols,
    __bf16* __restrict__ hfinal,     // (64, 512) base, cols dir*256..
    int sl) {
  extern __shared__ char smem[];
  __bf16* h_lds = (__bf16*)smem;          // 64*256
  __bf16* r_lds = h_lds + 64 * 256;
  __bf16* z_lds = r_lds + 64 * 256;

  int dir = blockIdx.x;
  gi += (size_t)dir * sl * BS * H3;
  Whh += (size_t)dir * H3 * NH;
  bhh += dir * H3;
  int ycol0 = dir * NH;

  for (int i = threadIdx.x; i < BS * NH; i += blockDim.x) h_lds[i] = (__bf16)0.0f;
  __syncthreads();

  int wave = threadIdx.x >> 5, lane = threadIdx.x & 31;
  int lm = lane & 15, lh = lane >> 4;

  for (int t = 0; t < sl; ++t) {
    int s = dir ? (sl - 1 - t) : t;
    const float* gi_s = gi + (size_t)s * BS * H3;

    // Phase 1: r & z gates, N in [0, 512): 128 tiles, 16 per wave.
    for (int ti = 0; ti < 16; ++ti) {
      int tile = wave * 16 + ti;
      int tm = (tile & 3) * 16;
      int tn = (tile >> 2) * 16;
      v8f acc = {};
#pragma unroll
      for (int kb = 0; kb < NH; kb += 32) {
        v16bf a = load_a_frag(h_lds, NH, tm + lm, kb, lh);
        v16bf b = load_b_frag(Whh, NH, tn + lm, kb, lh);
        acc = wmma_bf16(a, b, acc);
      }
      int n = tn + lm;                 // gate column in [0,512)
      bool isr = n < NH;
      int j = isr ? n : n - NH;
      float bb = bhh[n];
      __bf16* dst = isr ? r_lds : z_lds;
#pragma unroll
      for (int q = 0; q < 8; ++q) {
        int m = tm + q + 8 * lh;
        float g = acc[q] + bb + gi_s[m * H3 + n];
        dst[m * NH + j] = (__bf16)sigmoidf_(g);
      }
    }
    __syncthreads();

    // Phase 2: candidate gate + state update, N in [512, 768): 8 tiles/wave.
    float hnew[8][8];
    for (int ti = 0; ti < 8; ++ti) {
      int tile = wave * 8 + ti;
      int tm = (tile & 3) * 16;
      int tn = 2 * NH + (tile >> 2) * 16;
      v8f acc = {};
#pragma unroll
      for (int kb = 0; kb < NH; kb += 32) {
        v16bf a = load_a_frag(h_lds, NH, tm + lm, kb, lh);
        v16bf b = load_b_frag(Whh, NH, tn + lm, kb, lh);
        acc = wmma_bf16(a, b, acc);
      }
      int n = tn + lm;
      int j = n - 2 * NH;
      float bb = bhh[n];
#pragma unroll
      for (int q = 0; q < 8; ++q) {
        int m = tm + q + 8 * lh;
        float r = (float)r_lds[m * NH + j];
        float z = (float)z_lds[m * NH + j];
        float cand = tanhf(gi_s[m * H3 + n] + bb + r * acc[q]);
        float h = (float)h_lds[m * NH + j];
        hnew[ti][q] = (1.0f - z) * cand + z * h;
      }
    }
    __syncthreads();  // all WMMA reads of h for this step are done

    for (int ti = 0; ti < 8; ++ti) {
      int tile = wave * 8 + ti;
      int tm = (tile & 3) * 16;
      int j = (tile >> 2) * 16 + lm;
#pragma unroll
      for (int q = 0; q < 8; ++q) {
        int m = tm + q + 8 * lh;
        __bf16 hb = (__bf16)hnew[ti][q];
        h_lds[m * NH + j] = hb;
        size_t yo = ((size_t)s * BS + m) * ycols + ycol0 + j;
        ybf16[yo] = hb;
        if (yf32) yf32[yo] = hnew[ti][q];
      }
    }
    __syncthreads();
  }

  for (int i = threadIdx.x; i < BS * NH; i += blockDim.x) {
    int m = i / NH, j = i % NH;
    hfinal[m * (2 * NH) + ycol0 + j] = h_lds[m * NH + j];
  }
}

// ---------------------------------------------------------------------------
// Prep / pointwise kernels
// ---------------------------------------------------------------------------
// f32 (Nr x K) -> bf16 (NrP x Kp), zero-padded in both dims.
__global__ void k_conv_pad(const float* __restrict__ s, __bf16* __restrict__ d,
                           int Nr, int K, int Kp, int total) {
  int i = blockIdx.x * 256 + threadIdx.x;
  if (i >= total) return;
  int n = i / Kp, k = i % Kp;
  d[i] = (n < Nr && k < K) ? (__bf16)s[(size_t)n * K + k] : (__bf16)0.0f;
}

// W1 (512,300) K x N  ->  (EDNP, 512) N x K bf16 zero-padded rows.
__global__ void k_transpose_w1(const float* __restrict__ s,
                               __bf16* __restrict__ d) {
  int i = blockIdx.x * 256 + threadIdx.x;
  if (i >= EDNP * 2 * NH) return;
  int n = i / (2 * NH), k = i % (2 * NH);
  d[i] = (n < ED) ? (__bf16)s[(size_t)k * ED + n] : (__bf16)0.0f;
}

__global__ void k_zero_bf16(__bf16* __restrict__ p, int n) {
  int i = blockIdx.x * 256 + threadIdx.x;
  if (i < n) p[i] = (__bf16)0.0f;
}

// Encoder embedding gather -> (sl*64, EDP) bf16 with zero pad columns.
__global__ void k_embed_enc(const float* __restrict__ emb,
                            const int* __restrict__ inp,
                            __bf16* __restrict__ out, int total) {
  int i = blockIdx.x * 256 + threadIdx.x;
  if (i >= total) return;
  int row = i / EDP, d = i % EDP;
  out[i] = (d < ED) ? (__bf16)emb[(size_t)inp[row] * ED + d] : (__bf16)0.0f;
}

// Decoder embedding gather -> xcat[:, 0:300] (stride XCATWP).
__global__ void k_embed_dec(const float* __restrict__ emb,
                            const int* __restrict__ tok,
                            __bf16* __restrict__ xcat) {
  int i = blockIdx.x * 256 + threadIdx.x;
  if (i >= BS * ED) return;
  int b = i / ED, d = i % ED;
  xcat[b * XCATWP + d] = (__bf16)emb[(size_t)tok[b] * ED + d];
}

__global__ void k_init_dec(int* p) {
  if (threadIdx.x < BS) p[threadIdx.x] = 0;
}

// Attention: scores over seq, softmax, context -> xcat[:, 300:832] (bf16,
// zero-padding cols [812,832)).
__global__ __launch_bounds__(128) void k_attention(
    const float* __restrict__ w1e,     // (sl, 64, 300)
    const float* __restrict__ w2h,     // (64, 300)
    const float* __restrict__ Vatt,    // (300)
    const float* __restrict__ enc_out, // (sl, 64, 512)
    __bf16* __restrict__ xcat,         // (64, XCATWP)
    int sl) {
  int b = blockIdx.x;
  __shared__ float sw2h[ED];
  __shared__ float sc[512];
  for (int d = threadIdx.x; d < ED; d += blockDim.x) sw2h[d] = w2h[b * ED + d];
  __syncthreads();
  for (int s = threadIdx.x; s < sl; s += blockDim.x) {
    const float* row = w1e + ((size_t)s * BS + b) * ED;
    float acc = 0.0f;
    for (int d = 0; d < ED; ++d) acc += tanhf(row[d] + sw2h[d]) * Vatt[d];
    sc[s] = acc;
  }
  __syncthreads();
  if (threadIdx.x == 0) {
    float mx = sc[0];
    for (int s = 1; s < sl; ++s) mx = fmaxf(mx, sc[s]);
    float sum = 0.0f;
    for (int s = 0; s < sl; ++s) { float e = __expf(sc[s] - mx); sc[s] = e; sum += e; }
    float inv = 1.0f / sum;
    for (int s = 0; s < sl; ++s) sc[s] *= inv;
  }
  __syncthreads();
  for (int j = threadIdx.x; j < XCATWP - ED; j += blockDim.x) {
    float acc = 0.0f;
    if (j < 2 * NH) {
      for (int s = 0; s < sl; ++s)
        acc += sc[s] * enc_out[((size_t)s * BS + b) * (2 * NH) + j];
    }
    xcat[b * XCATWP + ED + j] = (__bf16)acc;
  }
}

// GRU gate combine: gi, gh (64, 900) f32; h (64,300) f32.
// Writes h' f32 (stride 300) and bf16 (stride EDP, zero pad cols).
__global__ void k_gru_gate(const float* __restrict__ gi,
                           const float* __restrict__ gh,
                           const float* __restrict__ h,
                           float* __restrict__ hn_f,
                           __bf16* __restrict__ hn_b) {
  int i = blockIdx.x * 256 + threadIdx.x;
  if (i >= BS * EDP) return;
  int b = i / EDP, j = i % EDP;
  if (j < ED) {
    const float* gib = gi + b * D3;
    const float* ghb = gh + b * D3;
    float r = sigmoidf_(gib[j] + ghb[j]);
    float z = sigmoidf_(gib[ED + j] + ghb[ED + j]);
    float cand = tanhf(gib[2 * ED + j] + r * ghb[2 * ED + j]);
    float hp = (1.0f - z) * cand + z * h[b * ED + j];
    hn_f[b * ED + j] = hp;
    hn_b[b * EDP + j] = (__bf16)hp;
  } else {
    hn_b[b * EDP + j] = (__bf16)0.0f;
  }
}

// Row-wise argmax over V=32000 (first-index tie break).
__global__ __launch_bounds__(256) void k_argmax(const float* __restrict__ logits,
                                                int* __restrict__ nxt) {
  int b = blockIdx.x;
  const float* row = logits + (size_t)b * VDEC;
  __shared__ float bv[256];
  __shared__ int bi[256];
  float best = -3.4e38f; int bidx = 0;
  for (int j = threadIdx.x; j < VDEC; j += 256) {
    float v = row[j];
    if (v > best || (v == best && j < bidx)) { best = v; bidx = j; }
  }
  bv[threadIdx.x] = best; bi[threadIdx.x] = bidx;
  __syncthreads();
  for (int s = 128; s > 0; s >>= 1) {
    if (threadIdx.x < s) {
      if (bv[threadIdx.x + s] > bv[threadIdx.x] ||
          (bv[threadIdx.x + s] == bv[threadIdx.x] &&
           bi[threadIdx.x + s] < bi[threadIdx.x])) {
        bv[threadIdx.x] = bv[threadIdx.x + s];
        bi[threadIdx.x] = bi[threadIdx.x + s];
      }
    }
    __syncthreads();
  }
  if (threadIdx.x == 0) nxt[b] = bi[0];
}

// ---------------------------------------------------------------------------
// Host orchestration
// ---------------------------------------------------------------------------
extern "C" void kernel_launch(void* const* d_in, const int* in_sizes, int n_in,
                              void* d_out, int out_size, void* d_ws, size_t ws_size,
                              hipStream_t stream) {
  (void)n_in; (void)out_size; (void)ws_size;
  const float* emb_enc  = (const float*)d_in[0];
  const float* Wih0     = (const float*)d_in[1];
  const float* Whh0     = (const float*)d_in[2];
  const float* bih0     = (const float*)d_in[3];
  const float* bhh0     = (const float*)d_in[4];
  const float* Wih1     = (const float*)d_in[5];
  const float* Whh1     = (const float*)d_in[6];
  const float* bih1     = (const float*)d_in[7];
  const float* bhh1     = (const float*)d_in[8];
  const float* Wout     = (const float*)d_in[9];
  const float* emb_dec  = (const float*)d_in[10];
  const float* dWih     = (const float*)d_in[11];
  const float* dWhh     = (const float*)d_in[12];
  const float* dbih     = (const float*)d_in[13];
  const float* dbhh     = (const float*)d_in[14];
  const float* W1       = (const float*)d_in[15];
  const float* l2_W     = (const float*)d_in[16];
  const float* l2_b     = (const float*)d_in[17];
  const float* l3_W     = (const float*)d_in[18];
  const float* l3_b     = (const float*)d_in[19];
  const float* V_att    = (const float*)d_in[20];
  const float* out_b    = (const float*)d_in[21];
  const int*   inp      = (const int*)d_in[22];

  const int sl = in_sizes[22] / BS;  // 128
  float* out = (float*)d_out;

  // --- bump allocator on d_ws --------------------------------------------
  size_t cur = 0;
  auto alloc = [&](size_t bytes) -> void* {
    size_t a = (cur + 255) & ~(size_t)255;
    cur = a + bytes;
    return (void*)((char*)d_ws + a);
  };
  // weights (bf16, padded)
  __bf16* wih0_b = (__bf16*)alloc((size_t)2 * H3 * EDP * 2);      // (2,768,320)
  __bf16* whh0_b = (__bf16*)alloc((size_t)2 * H3 * NH * 2);       // (2,768,256)
  __bf16* wih1_b = (__bf16*)alloc((size_t)2 * H3 * 2 * NH * 2);   // (2,768,512)
  __bf16* whh1_b = (__bf16*)alloc((size_t)2 * H3 * NH * 2);
  __bf16* wout_b = (__bf16*)alloc((size_t)EDNP * 2 * NH * 2);     // (384,512)
  __bf16* w1t_b  = (__bf16*)alloc((size_t)EDNP * 2 * NH * 2);     // (384,512)
  __bf16* l2w_b  = (__bf16*)alloc((size_t)EDNP * EDP * 2);        // (384,320)
  __bf16* l3w_b  = (__bf16*)alloc((size_t)EDNP * XCATWP * 2);     // (384,832)
  __bf16* dwih_b = (__bf16*)alloc((size_t)2 * D3P * EDP * 2);     // (2,1024,320)
  __bf16* dwhh_b = (__bf16*)alloc((size_t)2 * D3P * EDP * 2);
  __bf16* edec_b = (__bf16*)alloc((size_t)VDEC * EDP * 2);        // (32000,320)
  // activations
  __bf16* emb_b  = (__bf16*)alloc((size_t)sl * BS * EDP * 2);     // (8192,320)
  float*  gi     = (float*)alloc((size_t)2 * sl * BS * H3 * 4);   // reused L0/L1
  __bf16* x1_b   = (__bf16*)alloc((size_t)sl * BS * 2 * NH * 2);  // (8192,512)
  float*  eo_f   = (float*)alloc((size_t)sl * BS * 2 * NH * 4);
  __bf16* eo_b   = (__bf16*)alloc((size_t)sl * BS * 2 * NH * 2);
  __bf16* hcat_b = (__bf16*)alloc((size_t)2 * BS * 2 * NH * 2);   // (128,512)
  float*  w1e    = (float*)alloc((size_t)sl * BS * ED * 4);
  float*  hf[2]  = { (float*)alloc((size_t)2 * BS * ED * 4),
                     (float*)alloc((size_t)2 * BS * ED * 4) };
  __bf16* hb[2]  = { (__bf16*)alloc((size_t)2 * BS * EDP * 2),    // (128,320)
                     (__bf16*)alloc((size_t)2 * BS * EDP * 2) };
  float*  w2h    = (float*)alloc((size_t)BS * ED * 4);
  __bf16* xcat_b = (__bf16*)alloc((size_t)BS * XCATWP * 2);       // (64,832)
  float*  x_f    = (float*)alloc((size_t)BS * ED * 4);
  __bf16* x_b    = (__bf16*)alloc((size_t)BS * EDP * 2);          // (64,320)
  float*  gibuf  = (float*)alloc((size_t)BS * D3 * 4);
  float*  ghbuf  = (float*)alloc((size_t)BS * D3 * 4);
  int*    dtok[2]= { (int*)alloc(BS * 4), (int*)alloc(BS * 4) };

  auto convp = [&](const float* s, __bf16* d, int Nr, int K, int NrP, int Kp) {
    int total = NrP * Kp;
    k_conv_pad<<<(total + 255) / 256, 256, 0, stream>>>(s, d, Nr, K, Kp, total);
  };
  auto gemm = [&](const __bf16* A, int lda, const __bf16* Bt, int ldb,
                  const float* bias, float* C, int ldc,
                  __bf16* Cbf, int ldcb, int M, int N, int Kp) {
    dim3 g((N + 127) / 128, M / 32);
    wmma_gemm_nt<<<g, 256, 0, stream>>>(A, lda, Bt, ldb, bias, C, ldc, Cbf, ldcb,
                                        M, N, Kp);
  };
  auto zero_bf = [&](__bf16* p, int n) {
    k_zero_bf16<<<(n + 255) / 256, 256, 0, stream>>>(p, n);
  };

  // --- weight conversion (with zero padding) ------------------------------
  convp(Wih0, wih0_b, 2 * H3, ED, 2 * H3, EDP);
  convp(Whh0, whh0_b, 2 * H3, NH, 2 * H3, NH);
  convp(Wih1, wih1_b, 2 * H3, 2 * NH, 2 * H3, 2 * NH);
  convp(Whh1, whh1_b, 2 * H3, NH, 2 * H3, NH);
  convp(Wout, wout_b, ED, 2 * NH, EDNP, 2 * NH);
  convp(l2_W, l2w_b, ED, ED, EDNP, EDP);
  convp(l3_W, l3w_b, ED, XCATW, EDNP, XCATWP);
  for (int d = 0; d < 2; ++d) {
    convp(dWih + (size_t)d * D3 * ED, dwih_b + (size_t)d * D3P * EDP,
          D3, ED, D3P, EDP);
    convp(dWhh + (size_t)d * D3 * ED, dwhh_b + (size_t)d * D3P * EDP,
          D3, ED, D3P, EDP);
  }
  convp(emb_dec, edec_b, VDEC, ED, VDEC, EDP);
  k_transpose_w1<<<(EDNP * 2 * NH + 255) / 256, 256, 0, stream>>>(W1, w1t_b);
  zero_bf(hb[0], 2 * BS * EDP);
  zero_bf(hb[1], 2 * BS * EDP);
  zero_bf(x_b, BS * EDP);

  // --- encoder ------------------------------------------------------------
  int embtot = sl * BS * EDP;
  k_embed_enc<<<(embtot + 255) / 256, 256, 0, stream>>>(emb_enc, inp, emb_b, embtot);

  const size_t giStride = (size_t)sl * BS * H3;
  for (int d = 0; d < 2; ++d)
    gemm(emb_b, EDP, wih0_b + (size_t)d * H3 * EDP, EDP, bih0 + d * H3,
         gi + d * giStride, H3, nullptr, 0, sl * BS, H3, EDP);
  gru_scan<<<2, 256, 3 * BS * NH * 2, stream>>>(gi, whh0_b, bhh0, nullptr, x1_b,
                                                2 * NH, hcat_b, sl);
  for (int d = 0; d < 2; ++d)
    gemm(x1_b, 2 * NH, wih1_b + (size_t)d * H3 * 2 * NH, 2 * NH, bih1 + d * H3,
         gi + d * giStride, H3, nullptr, 0, sl * BS, H3, 2 * NH);
  gru_scan<<<2, 256, 3 * BS * NH * 2, stream>>>(gi, whh1_b, bhh1, eo_f, eo_b,
                                                2 * NH,
                                                hcat_b + (size_t)BS * 2 * NH, sl);

  // h = [hcat0; hcat1] @ Wout^T  -> (128, 300) f32 + bf16(padded)
  gemm(hcat_b, 2 * NH, wout_b, 2 * NH, nullptr, hf[0], ED, hb[0], EDP,
       2 * BS, ED, 2 * NH);
  // w1e = enc_out @ W1  (W1 pre-transposed to N x K)
  gemm(eo_b, 2 * NH, w1t_b, 2 * NH, nullptr, w1e, ED, nullptr, 0,
       sl * BS, ED, 2 * NH);
  k_init_dec<<<1, 64, 0, stream>>>(dtok[0]);

  // --- decoder: 32 unrolled steps with ping-pong state --------------------
  int c = 0, nx = 1;
  for (int t = 0; t < OUT_SL; ++t) {
    // w2h = h[-1] @ l2_W^T + l2_b
    gemm(hb[c] + (size_t)BS * EDP, EDP, l2w_b, EDP, l2_b, w2h, ED, nullptr, 0,
         BS, ED, EDP);
    k_attention<<<BS, 128, 0, stream>>>(w1e, w2h, V_att, eo_f, xcat_b, sl);
    k_embed_dec<<<(BS * ED + 255) / 256, 256, 0, stream>>>(emb_dec, dtok[c], xcat_b);
    // x = [e | Xa] @ l3_W^T + l3_b
    gemm(xcat_b, XCATWP, l3w_b, XCATWP, l3_b, x_f, ED, x_b, EDP,
         BS, ED, XCATWP);
    // cell 0
    gemm(x_b, EDP, dwih_b, EDP, dbih, gibuf, D3, nullptr, 0, BS, D3, EDP);
    gemm(hb[c], EDP, dwhh_b, EDP, dbhh, ghbuf, D3, nullptr, 0, BS, D3, EDP);
    k_gru_gate<<<(BS * EDP + 255) / 256, 256, 0, stream>>>(gibuf, ghbuf, hf[c],
                                                           hf[nx], hb[nx]);
    // cell 1 (input = h0n)
    gemm(hb[nx], EDP, dwih_b + (size_t)D3P * EDP, EDP, dbih + D3, gibuf, D3,
         nullptr, 0, BS, D3, EDP);
    gemm(hb[c] + (size_t)BS * EDP, EDP, dwhh_b + (size_t)D3P * EDP, EDP,
         dbhh + D3, ghbuf, D3, nullptr, 0, BS, D3, EDP);
    k_gru_gate<<<(BS * EDP + 255) / 256, 256, 0, stream>>>(
        gibuf, ghbuf, hf[c] + (size_t)BS * ED,
        hf[nx] + (size_t)BS * ED, hb[nx] + (size_t)BS * EDP);
    // logits = h1n @ emb_dec^T + out_b  -> d_out slab t
    gemm(hb[nx] + (size_t)BS * EDP, EDP, edec_b, EDP, out_b,
         out + (size_t)t * BS * VDEC, VDEC, nullptr, 0, BS, VDEC, EDP);
    k_argmax<<<BS, 256, 0, stream>>>(out + (size_t)t * BS * VDEC, dtok[nx]);
    c ^= 1; nx ^= 1;
  }
}
